// DualSpectralAttention_54443005444515
// MI455X (gfx1250) — compile-verified, matrix operand order checked by reading, not run on hardware
//
#include <hip/hip_runtime.h>

typedef unsigned short u16;
typedef __attribute__((ext_vector_type(16))) __bf16 v16bf;
typedef __attribute__((ext_vector_type(8)))  float  v8f;

#define D_MODEL 1024
#define SEQN    2048
#define BATCH   2
#define NHEADS  16
#define HDIM    64
#define MROWS   (BATCH * SEQN)   // 4096

__device__ __forceinline__ u16 f2bf(float f) {
  unsigned u = __builtin_bit_cast(unsigned, f);
  u += 0x7FFFu + ((u >> 16) & 1u);   // round-to-nearest-even
  return (u16)(u >> 16);
}

struct B32x2 { uint4 lo, hi; };

// Build one 16x32 bf16 WMMA operand fragment from an LDS row.
// A-layout (ISA 7.12.2): lanes 0-15 hold K {0..7,16..23}, lanes 16-31 hold
// K {8..15,24..31}; caller passes p = &row[koff] with koff = (lane>>4)*8.
__device__ __forceinline__ v16bf make_frag(const u16* p) {
  B32x2 t;
  t.lo = *(const uint4*)(p);
  t.hi = *(const uint4*)(p + 16);
  return __builtin_bit_cast(v16bf, t);
}

// Async global->LDS 16B copy (CDNA5 GLOBAL_LOAD_ASYNC_TO_LDS_B128, ASYNCcnt).
// LDS dest = low 32 bits of the flat shared address (ISA 10.2 aperture rule).
__device__ __forceinline__ void async_copy_b128(void* lds_dst, const void* gsrc) {
  unsigned lds = (unsigned)(size_t)lds_dst;
  unsigned long long ga = (unsigned long long)(size_t)gsrc;
  asm volatile("global_load_async_to_lds_b128 %0, %1, off"
               :: "v"(lds), "v"(ga) : "memory");
}
__device__ __forceinline__ void async_wait0() {
  asm volatile("s_wait_asynccnt 0x0" ::: "memory");
}

// ---------------------------------------------------------------- fp32 -> bf16
__global__ __launch_bounds__(256)
void cvt_f32_bf16(const float* __restrict__ s, u16* __restrict__ d, int n4) {
  int i = blockIdx.x * 256 + threadIdx.x;
  if (i >= n4) return;
  float4 v = ((const float4*)s)[i];
  ushort4 o;
  o.x = f2bf(v.x); o.y = f2bf(v.y); o.z = f2bf(v.z); o.w = f2bf(v.w);
  ((ushort4*)d)[i] = o;
}

// ------------------------------------------- fold gate/W_cross/shift into Wq
// A[h] = sigmoid(gate[h]) * W_cross[h] + (1-g) * M[h],  M[h][f][g]=w[h,g-f]
// Wq_eff[h*64+g, :] = sum_f A[h][f][g] * Wq[h*64+f, :]   (bf16 out)
// bq_eff[h*64+g]    = sum_f bq[h*64+f] * A[h][f][g]
__global__ __launch_bounds__(256)
void build_q_eff(const float* __restrict__ Wq, const float* __restrict__ bq,
                 const float* __restrict__ Wcross, const float* __restrict__ shiftw,
                 const float* __restrict__ gate,
                 u16* __restrict__ wq_eff, float* __restrict__ bq_eff) {
  const int h = blockIdx.x;
  __shared__ float Al[64][65];
  const float gt = 1.0f / (1.0f + __expf(-gate[h]));
  for (int i = threadIdx.x; i < 64 * 64; i += 256) {
    int f = i >> 6, g = i & 63;
    int tau = g - f;
    float m = (tau >= 0 && tau < 32) ? shiftw[h * 32 + tau] : 0.0f;
    Al[f][g] = gt * Wcross[h * 4096 + i] + (1.0f - gt) * m;
  }
  __syncthreads();
  for (int g = threadIdx.x; g < 64; g += 256) {
    float s = 0.0f;
    for (int f = 0; f < 64; ++f) s += bq[h * 64 + f] * Al[f][g];
    bq_eff[h * 64 + g] = s;
  }
  const float* wqh = Wq + (long long)h * 64 * D_MODEL;
  for (int idx = threadIdx.x; idx < 64 * D_MODEL; idx += 256) {
    int g = idx >> 10, k = idx & (D_MODEL - 1);
    float s = 0.0f;
    for (int f = 0; f < 64; ++f) s += Al[f][g] * wqh[f * D_MODEL + k];
    wq_eff[((long long)h * 64 + g) * D_MODEL + k] = f2bf(s);
  }
}

// ------------------------------------------------- C[M,N] = A[M,K] @ B[N,K]^T
// bf16 in, fp32 accumulate, bf16 or fp32 out.  Block tile 128x128, BK=32,
// 8 wave32 waves in a 4(M) x 2(N) grid, wave tile 32x64 -> 8 WMMAs / K-step.
// Tiles are staged global->LDS with GLOBAL_LOAD_ASYNC_TO_LDS_B128 (no VGPR
// staging), synchronized with s_wait_asynccnt + workgroup barrier.
template <bool OUT_F32>
__global__ __launch_bounds__(256)
void gemm_bf16_nt(const u16* __restrict__ A, const u16* __restrict__ Bm,
                  const float* __restrict__ bias, void* __restrict__ C,
                  int M, int N, int K) {
  __shared__ alignas(16) u16 As[128][40];
  __shared__ alignas(16) u16 Bs[128][40];
  const int tid  = threadIdx.x;
  const int lane = tid & 31;
  const int wave = tid >> 5;
  const int ln   = lane & 15;
  const int koff = (lane >> 4) * 8;
  const int wm = (wave & 3) * 32;
  const int wn = (wave >> 2) * 64;
  const long long rowA0 = (long long)blockIdx.x * 128;
  const long long rowB0 = (long long)blockIdx.y * 128;

  const v8f zero8 = {0, 0, 0, 0, 0, 0, 0, 0};
  v8f acc[2][4];
#pragma unroll
  for (int m = 0; m < 2; ++m)
#pragma unroll
    for (int n = 0; n < 4; ++n) acc[m][n] = zero8;

  for (int kb = 0; kb < K; kb += 32) {
#pragma unroll
    for (int i = 0; i < 2; ++i) {               // 128 rows x 32 halves per tile
      int idx = tid + i * 256;                  // 512 16B chunks, 2 per thread
      int r = idx >> 2;
      int c = (idx & 3) * 8;
      async_copy_b128(&As[r][c], &A[(rowA0 + r) * K + kb + c]);
      async_copy_b128(&Bs[r][c], &Bm[(rowB0 + r) * K + kb + c]);
    }
    if (kb + 32 < K) {                          // gfx1250 global_prefetch_b8
      __builtin_prefetch(&A[(rowA0 + (tid >> 1)) * K + kb + 32], 0, 0);
      __builtin_prefetch(&Bm[(rowB0 + (tid >> 1)) * K + kb + 32], 0, 0);
    }
    async_wait0();
    __syncthreads();
    v16bf af[2], bfv[4];
#pragma unroll
    for (int m = 0; m < 2; ++m) af[m] = make_frag(&As[wm + m * 16 + ln][koff]);
#pragma unroll
    for (int n = 0; n < 4; ++n) bfv[n] = make_frag(&Bs[wn + n * 16 + ln][koff]);
#pragma unroll
    for (int m = 0; m < 2; ++m)
#pragma unroll
      for (int n = 0; n < 4; ++n)
        acc[m][n] = __builtin_amdgcn_wmma_f32_16x16x32_bf16(
            false, af[m], false, bfv[n], (short)0, acc[m][n], false, false);
    __syncthreads();
  }

  const int half8 = (lane >> 4) * 8;            // C layout: row = r + half*8
#pragma unroll
  for (int m = 0; m < 2; ++m)
#pragma unroll
    for (int n = 0; n < 4; ++n) {
      long long col = rowB0 + wn + n * 16 + ln;
      float bb = bias ? bias[col] : 0.0f;
#pragma unroll
      for (int r = 0; r < 8; ++r) {
        long long row = rowA0 + wm + m * 16 + half8 + r;
        float v = acc[m][n][r] + bb;
        if (OUT_F32) ((float*)C)[row * N + col] = v;
        else         ((u16*)C)[row * N + col] = f2bf(v);
      }
    }
}

// --------------------------------------------- causal flash attention, bf16
// grid.x = q-tile (S/64), grid.y = b*H + h.  128 threads = 4 waves; wave w
// owns q rows [w*16, w*16+16) -> wave-local online softmax.
__global__ __launch_bounds__(128)
void attn_causal(const u16* __restrict__ Q, const u16* __restrict__ Kg,
                 const u16* __restrict__ Vg, u16* __restrict__ O) {
  const int qt = blockIdx.x;
  const int bh = blockIdx.y;
  const int b = bh >> 4, h = bh & 15;
  const int tid = threadIdx.x;
  const int lane = tid & 31, wave = tid >> 5;
  const int ln = lane & 15;
  const int halfsel = lane >> 4;
  const int koffB = halfsel * 8;

  __shared__ alignas(16) u16 Qs[64][72];
  __shared__ alignas(16) u16 Ks[64][72];
  __shared__ alignas(16) u16 Vt[64][72];   // V transposed: [d][k]
  __shared__ alignas(16) u16 Ps[64][72];   // per-wave staging of P tiles

  const long long qbase = ((long long)(b * SEQN + qt * 64)) * D_MODEL + h * HDIM;
#pragma unroll
  for (int i = 0; i < 4; ++i) {
    int idx = tid + i * 128;
    int r = idx >> 3;
    int c = (idx & 7) * 8;
    async_copy_b128(&Qs[r][c], &Q[qbase + (long long)r * D_MODEL + c]);
  }
  async_wait0();
  __syncthreads();

  v16bf qf[2];
#pragma unroll
  for (int ks = 0; ks < 2; ++ks)
    qf[ks] = make_frag(&Qs[wave * 16 + ln][ks * 32 + koffB]);

  const v8f zero8 = {0, 0, 0, 0, 0, 0, 0, 0};
  v8f oacc[4] = {zero8, zero8, zero8, zero8};
  float mrow[8], lrow[8];
#pragma unroll
  for (int r = 0; r < 8; ++r) { mrow[r] = -1e30f; lrow[r] = 0.0f; }
  const float scale = 0.125f;  // 64^-0.5

  for (int kt = 0; kt <= qt; ++kt) {
    const long long kvbase =
        ((long long)(b * SEQN + kt * 64)) * D_MODEL + h * HDIM;
#pragma unroll
    for (int i = 0; i < 4; ++i) {
      int idx = tid + i * 128;
      int r = idx >> 3;
      int c = (idx & 7) * 8;
      // K tile: direct async global->LDS (no VGPR staging)
      async_copy_b128(&Ks[r][c], &Kg[kvbase + (long long)r * D_MODEL + c]);
      // V tile: VGPR path (element-wise transpose into LDS)
      union { uint4 q; u16 s[8]; } u;
      u.q = *(const uint4*)&Vg[kvbase + (long long)r * D_MODEL + c];
#pragma unroll
      for (int j = 0; j < 8; ++j) Vt[c + j][r] = u.s[j];
    }
    async_wait0();
    __syncthreads();

    // S = Q K^T  (16 q-rows x 64 k-cols per wave)
    v8f sacc[4] = {zero8, zero8, zero8, zero8};
#pragma unroll
    for (int ks = 0; ks < 2; ++ks)
#pragma unroll
      for (int n = 0; n < 4; ++n) {
        v16bf kf = make_frag(&Ks[n * 16 + ln][ks * 32 + koffB]);
        sacc[n] = __builtin_amdgcn_wmma_f32_16x16x32_bf16(
            false, qf[ks], false, kf, (short)0, sacc[n], false, false);
      }

    // scale + causal mask (only bites on the diagonal tile)
    const int qrow0 = qt * 64 + wave * 16 + halfsel * 8;
#pragma unroll
    for (int n = 0; n < 4; ++n) {
      const int kcol = kt * 64 + n * 16 + ln;
#pragma unroll
      for (int r = 0; r < 8; ++r) {
        float s = sacc[n][r] * scale;
        if (kcol > qrow0 + r) s = -1e30f;
        sacc[n][r] = s;
      }
    }

    // online softmax (rows live in 16-lane halves -> xor-reduce 8,4,2,1)
    float alpha[8];
#pragma unroll
    for (int r = 0; r < 8; ++r) {
      float mx = fmaxf(fmaxf(sacc[0][r], sacc[1][r]),
                       fmaxf(sacc[2][r], sacc[3][r]));
#pragma unroll
      for (int off = 8; off >= 1; off >>= 1)
        mx = fmaxf(mx, __shfl_xor(mx, off, 32));
      float mn = fmaxf(mrow[r], mx);
      alpha[r] = __expf(mrow[r] - mn);
      mrow[r] = mn;
    }
    float rs[8];
#pragma unroll
    for (int r = 0; r < 8; ++r) rs[r] = 0.0f;
#pragma unroll
    for (int n = 0; n < 4; ++n)
#pragma unroll
      for (int r = 0; r < 8; ++r) {
        float p = __expf(sacc[n][r] - mrow[r]);
        rs[r] += p;
        Ps[wave * 16 + halfsel * 8 + r][n * 16 + ln] = f2bf(p);
      }
#pragma unroll
    for (int r = 0; r < 8; ++r) {
      float t = rs[r];
#pragma unroll
      for (int off = 8; off >= 1; off >>= 1) t += __shfl_xor(t, off, 32);
      lrow[r] = lrow[r] * alpha[r] + t;
#pragma unroll
      for (int n = 0; n < 4; ++n) oacc[n][r] *= alpha[r];
    }

    // O += P V  (P staged via per-wave LDS region; same-wave DS ops are in
    // order per ISA so no barrier needed between the scalar stores and frags)
    v16bf pf[2];
#pragma unroll
    for (int ks = 0; ks < 2; ++ks)
      pf[ks] = make_frag(&Ps[wave * 16 + ln][ks * 32 + koffB]);
#pragma unroll
    for (int n = 0; n < 4; ++n)
#pragma unroll
      for (int ks = 0; ks < 2; ++ks) {
        v16bf vf = make_frag(&Vt[n * 16 + ln][ks * 32 + koffB]);
        oacc[n] = __builtin_amdgcn_wmma_f32_16x16x32_bf16(
            false, pf[ks], false, vf, (short)0, oacc[n], false, false);
      }
    __syncthreads();   // protect Ks/Vt before next iteration's overwrite
  }

  const long long obase =
      ((long long)(b * SEQN + qt * 64 + wave * 16 + halfsel * 8)) * D_MODEL +
      h * HDIM;
#pragma unroll
  for (int r = 0; r < 8; ++r) {
    float inv = 1.0f / lrow[r];
#pragma unroll
    for (int n = 0; n < 4; ++n)
      O[obase + (long long)r * D_MODEL + n * 16 + ln] = f2bf(oacc[n][r] * inv);
  }
}

// ------------------------------------------------------------------ launcher
extern "C" void kernel_launch(void* const* d_in, const int* in_sizes, int n_in,
                              void* d_out, int out_size, void* d_ws, size_t ws_size,
                              hipStream_t stream) {
  (void)in_sizes; (void)n_in; (void)out_size; (void)ws_size;
  const float* x      = (const float*)d_in[0];
  const float* Wq     = (const float*)d_in[1];
  const float* bq     = (const float*)d_in[2];
  const float* Wk     = (const float*)d_in[3];
  const float* bk     = (const float*)d_in[4];
  const float* Wv     = (const float*)d_in[5];
  const float* bv     = (const float*)d_in[6];
  const float* Wo     = (const float*)d_in[7];
  const float* bo     = (const float*)d_in[8];
  const float* Wcross = (const float*)d_in[9];
  const float* shiftw = (const float*)d_in[10];
  const float* gate   = (const float*)d_in[11];

  char* ws = (char*)d_ws;
  u16*   x_bf   = (u16*)(ws);                            //  8 MB
  u16*   wq_eff = (u16*)(ws + (8ull  << 20));            //  2 MB
  u16*   wk_bf  = (u16*)(ws + (10ull << 20));            //  2 MB
  u16*   wv_bf  = (u16*)(ws + (12ull << 20));            //  2 MB
  u16*   wo_bf  = (u16*)(ws + (14ull << 20));            //  2 MB
  u16*   Qb     = (u16*)(ws + (16ull << 20));            //  8 MB
  u16*   Kb     = (u16*)(ws + (24ull << 20));            //  8 MB
  u16*   Vb     = (u16*)(ws + (32ull << 20));            //  8 MB
  u16*   Ob     = (u16*)(ws + (40ull << 20));            //  8 MB
  float* bq_eff = (float*)(ws + (48ull << 20));          //  4 KB

  const int nX4 = MROWS * D_MODEL / 4;     // 1048576
  const int nW4 = D_MODEL * D_MODEL / 4;   //  262144
  cvt_f32_bf16<<<(nX4 + 255) / 256, 256, 0, stream>>>(x,  x_bf,  nX4);
  cvt_f32_bf16<<<(nW4 + 255) / 256, 256, 0, stream>>>(Wk, wk_bf, nW4);
  cvt_f32_bf16<<<(nW4 + 255) / 256, 256, 0, stream>>>(Wv, wv_bf, nW4);
  cvt_f32_bf16<<<(nW4 + 255) / 256, 256, 0, stream>>>(Wo, wo_bf, nW4);

  build_q_eff<<<NHEADS, 256, 0, stream>>>(Wq, bq, Wcross, shiftw, gate,
                                          wq_eff, bq_eff);

  dim3 gemm_grid(MROWS / 128, D_MODEL / 128);
  gemm_bf16_nt<false><<<gemm_grid, 256, 0, stream>>>(
      x_bf, wq_eff, bq_eff, Qb, MROWS, D_MODEL, D_MODEL);
  gemm_bf16_nt<false><<<gemm_grid, 256, 0, stream>>>(
      x_bf, wk_bf, bk, Kb, MROWS, D_MODEL, D_MODEL);
  gemm_bf16_nt<false><<<gemm_grid, 256, 0, stream>>>(
      x_bf, wv_bf, bv, Vb, MROWS, D_MODEL, D_MODEL);

  dim3 attn_grid(SEQN / 64, BATCH * NHEADS);
  attn_causal<<<attn_grid, 128, 0, stream>>>(Qb, Kb, Vb, Ob);

  gemm_bf16_nt<true><<<gemm_grid, 256, 0, stream>>>(
      Ob, wo_bf, bo, (float*)d_out, MROWS, D_MODEL, D_MODEL);
}
